// GPSDecoder_53395033424436
// MI455X (gfx1250) — matrix operand order repaired and checked by reading, not compiled, not run on hardware
//
#include <hip/hip_runtime.h>
#include <hip/hip_bf16.h>

typedef __attribute__((ext_vector_type(16))) _Float16 v16h;
typedef __attribute__((ext_vector_type(8)))  _Float16 v8h;
typedef __attribute__((ext_vector_type(8)))  float    v8f;
typedef __attribute__((ext_vector_type(4)))  float    v4f;

#define ZD    128
#define HDIM  256
#define LNUM  4
#define BB    64
#define NPGC  64
#define EPGC  1024
#define NN    (BB * NPGC)        // 4096
#define EE    (BB * EPGC)        // 65536
#define PPC   (NPGC * (NPGC-1))  // 4032

// ---- output offsets (floats), reference return order ----
#define OFF_CEN 0                       // (N,3)      12288
#define OFF_MAT 12288                   // (N,32)     131072
#define OFF_NT  143360                  // (N,3)      12288
#define OFF_FR  155648                  // (N,1)      4096
#define OFF_EA  159744                  // (E,4)      262144
#define OFF_AL  421888                  // (B*P)      258048
#define OFF_AW  679936
#define OFF_TE  937984
#define OFF_TW  1196032

// =====================================================================
// Weight transpose: WT[n*K + k] = W[k*N + n]   (B-fragments become
// two contiguous 8-float runs per lane -> global_load_b128 streams)
// =====================================================================
__global__ void k_transpose(const float* __restrict__ W, float* __restrict__ WT,
                            int K, int N) {
  const int i = blockIdx.x * blockDim.x + threadIdx.x;
  if (i >= K * N) return;
  const int k = i / N, n = i - k * N;
  WT[(size_t)n * K + k] = W[i];
}

// =====================================================================
// WMMA GEMM, N fixed to 256:  C[M,256] = act(A[M,K] @ W[K,256] + bias)
// WT is the transposed weight (256 x K). One 16x16 tile per wave.
// MODE 0: A row m = A0 + m*K            (ksplit is ignored)
// MODE 1: col<ksplit -> A0[idx0[m]] row, else A1[idx1[m]] row (lda=HDIM)
// MODE 2: col<256 -> node_queries[(m&63)], else z_cond[(m>>6)]
// ksplit is a multiple of 32, so segment choice is uniform per K-tile.
// =====================================================================
template <int MODE>
__global__ __launch_bounds__(128)
void k_gemm(const float* __restrict__ A0, const float* __restrict__ A1,
            const int* __restrict__ idx0, const int* __restrict__ idx1,
            int ksplit,
            const float* __restrict__ WT, const float* __restrict__ bias,
            float* __restrict__ C, int M, int K, int relu) {
  const int N = 256;
  const int wave = threadIdx.x >> 5;
  const int lane = threadIdx.x & 31;
  const int tiles_n = N >> 4;
  const int tile = blockIdx.x * 4 + wave;
  if (tile >= (M >> 4) * tiles_n) return;          // wave-uniform
  const int tm = tile / tiles_n;
  const int tn = tile - tm * tiles_n;
  const int lr = lane & 15;
  const int hi = lane >> 4;
  const int arow = tm * 16 + lr;
  const int ncol = tn * 16 + lr;

  const float* base0;
  const float* base1;
  if (MODE == 0) {
    base0 = A0 + (size_t)arow * K;
    base1 = base0;
  } else if (MODE == 1) {
    base0 = A0 + (size_t)idx0[arow] * HDIM;
    base1 = A1 + (size_t)idx1[arow] * HDIM - ksplit;
  } else {
    base0 = A0 + (size_t)(arow & 63) * 256;
    base1 = A1 + (size_t)(arow >> 6) * 256 - 256;
  }
  const float* wrow = WT + (size_t)ncol * K;

  v8f acc = {};
  for (int k0 = 0; k0 < K; k0 += 32) {
    // A fragment (16x32 f16): lanes 0-15 K=k0+{0..7,16..23}; lanes 16-31 +8.
    const float* ap = (((MODE == 0) || (k0 < ksplit)) ? base0 : base1)
                      + k0 + (hi << 3);
    const v4f a0 = *(const v4f*)(ap);
    const v4f a1 = *(const v4f*)(ap + 4);
    const v4f a2 = *(const v4f*)(ap + 16);
    const v4f a3 = *(const v4f*)(ap + 20);
    v16h a;
#pragma unroll
    for (int j = 0; j < 4; ++j) {
      a[j]      = (_Float16)a0[j];
      a[j + 4]  = (_Float16)a1[j];
      a[j + 8]  = (_Float16)a2[j];
      a[j + 12] = (_Float16)a3[j];
    }
    // B fragment (32x16 f16): lane col = ncol, K = k0+{0..7,16..23}; lanes 16-31 +8
    const float* wp = wrow + k0 + (hi << 3);
    __builtin_prefetch(wp + 64, 0, 1);             // global_prefetch_b8
    const v4f b0 = *(const v4f*)(wp);
    const v4f b1 = *(const v4f*)(wp + 4);
    const v4f b2 = *(const v4f*)(wp + 16);
    const v4f b3 = *(const v4f*)(wp + 20);
    v16h bf;
#pragma unroll
    for (int j = 0; j < 4; ++j) {
      bf[j]      = (_Float16)b0[j];
      bf[j + 4]  = (_Float16)b1[j];
      bf[j + 8]  = (_Float16)b2[j];
      bf[j + 12] = (_Float16)b3[j];
    }
    acc = __builtin_amdgcn_wmma_f32_16x16x32_f16(false, a, false, bf,
                                                 (short)0, acc, false, false);
  }
  const float bv = bias ? bias[ncol] : 0.0f;
#pragma unroll
  for (int g = 0; g < 8; ++g) {
    float v = acc[g] + bv;
    if (relu) v = fmaxf(v, 0.0f);
    const int m = hi ? (g + 8) : g;
    C[(size_t)(tm * 16 + m) * N + ncol] = v;
  }
}

// =====================================================================
// Fused pair-adjacency MLP: per wave, 16 pairs.
// pair = [h_i, h_j, h_i*h_j] (768) -> relu(@W1 768x256) -> relu(@W2 256x128)
//      -> @W3 128x2  -> adj_logits / adj_weights
// wt1 = W1^T (256x768), wt2 = W2^T (128x256)
// =====================================================================
__global__ __launch_bounds__(32)
void k_adj_fused(const float* __restrict__ h,
                 const float* __restrict__ wt1, const float* __restrict__ b1,
                 const float* __restrict__ wt2, const float* __restrict__ b2,
                 const float* __restrict__ w3, const float* __restrict__ b3,
                 float* __restrict__ out_logits, float* __restrict__ out_weights) {
  __shared__ _Float16 pf[16 * 768];   // pair features, f16
  __shared__ _Float16 x1[16 * 256];
  __shared__ _Float16 x2[16 * 128];
  __shared__ float    w3s[128 * 2];
  __shared__ float    b3s[2];

  const int wid  = blockIdx.x;            // 0 .. B*PPC/16 - 1   (PPC%16==0)
  const int b    = wid / (PPC / 16);
  const int pt   = wid % (PPC / 16);
  const int lane = threadIdx.x;
  const int lr   = lane & 15;
  const int hi   = lane >> 4;

  for (int t = lane; t < 256; t += 32) w3s[t] = w3[t];
  if (lane < 2) b3s[lane] = b3[lane];

  // stage pair features (f16) into LDS (vector f32 loads, packed converts)
  for (int rr = 0; rr < 16; ++rr) {
    const int p  = pt * 16 + rr;
    const int ii = p / 63;
    const int kk = p % 63;
    const int jj = kk + (kk >= ii ? 1 : 0);
    const float* hi_row = h + (size_t)(b * NPGC + ii) * HDIM;
    const float* hj_row = h + (size_t)(b * NPGC + jj) * HDIM;
    for (int c0 = lane * 4; c0 < HDIM; c0 += 128) {
      const v4f va = *(const v4f*)(hi_row + c0);
      const v4f vb = *(const v4f*)(hj_row + c0);
#pragma unroll
      for (int j = 0; j < 4; ++j) {
        pf[rr * 768 + c0 + j]       = (_Float16)va[j];
        pf[rr * 768 + 256 + c0 + j] = (_Float16)vb[j];
        pf[rr * 768 + 512 + c0 + j] = (_Float16)(va[j] * vb[j]);
      }
    }
  }
  __syncthreads();

  // ---- layer 1: 16x768 @ 768x256, bias+relu -> x1 (f16 in LDS) ----
  for (int n = 0; n < 16; ++n) {
    v8f acc = {};
    const int ncol = n * 16 + lr;
    const float* wrow = wt1 + (size_t)ncol * 768;
    for (int kt = 0; kt < 24; ++kt) {
      const int hb = lr * 768 + kt * 32 + (hi << 3);
      const v8h lo  = *(const v8h*)&pf[hb];
      const v8h hi8 = *(const v8h*)&pf[hb + 16];
      v16h a;
#pragma unroll
      for (int j = 0; j < 8; ++j) { a[j] = lo[j]; a[j + 8] = hi8[j]; }
      const float* wp = wrow + kt * 32 + (hi << 3);
      __builtin_prefetch(wp + 64, 0, 1);
      const v4f b0 = *(const v4f*)(wp);
      const v4f b1v = *(const v4f*)(wp + 4);
      const v4f b2v = *(const v4f*)(wp + 16);
      const v4f b3v = *(const v4f*)(wp + 20);
      v16h bf;
#pragma unroll
      for (int j = 0; j < 4; ++j) {
        bf[j]      = (_Float16)b0[j];
        bf[j + 4]  = (_Float16)b1v[j];
        bf[j + 8]  = (_Float16)b2v[j];
        bf[j + 12] = (_Float16)b3v[j];
      }
      acc = __builtin_amdgcn_wmma_f32_16x16x32_f16(false, a, false, bf,
                                                   (short)0, acc, false, false);
    }
    const float bv = b1[ncol];
#pragma unroll
    for (int g = 0; g < 8; ++g) {
      const int m = hi ? (g + 8) : g;
      x1[m * 256 + ncol] = (_Float16)fmaxf(acc[g] + bv, 0.0f);
    }
  }
  __syncthreads();

  // ---- layer 2: 16x256 @ 256x128, bias+relu -> x2 (f16 in LDS) ----
  for (int n = 0; n < 8; ++n) {
    v8f acc = {};
    const int ncol = n * 16 + lr;
    const float* wrow = wt2 + (size_t)ncol * 256;
    for (int kt = 0; kt < 8; ++kt) {
      const int hb = lr * 256 + kt * 32 + (hi << 3);
      const v8h lo  = *(const v8h*)&x1[hb];
      const v8h hi8 = *(const v8h*)&x1[hb + 16];
      v16h a;
#pragma unroll
      for (int j = 0; j < 8; ++j) { a[j] = lo[j]; a[j + 8] = hi8[j]; }
      const float* wp = wrow + kt * 32 + (hi << 3);
      const v4f b0 = *(const v4f*)(wp);
      const v4f b1v = *(const v4f*)(wp + 4);
      const v4f b2v = *(const v4f*)(wp + 16);
      const v4f b3v = *(const v4f*)(wp + 20);
      v16h bf;
#pragma unroll
      for (int j = 0; j < 4; ++j) {
        bf[j]      = (_Float16)b0[j];
        bf[j + 4]  = (_Float16)b1v[j];
        bf[j + 8]  = (_Float16)b2v[j];
        bf[j + 12] = (_Float16)b3v[j];
      }
      acc = __builtin_amdgcn_wmma_f32_16x16x32_f16(false, a, false, bf,
                                                   (short)0, acc, false, false);
    }
    const float bv = b2[ncol];
#pragma unroll
    for (int g = 0; g < 8; ++g) {
      const int m = hi ? (g + 8) : g;
      x2[m * 128 + ncol] = (_Float16)fmaxf(acc[g] + bv, 0.0f);
    }
  }
  __syncthreads();

  // ---- layer 3: 16x128 @ 128x2 (VALU). lane = (row, col) ----
  const int row = lr, col = hi;
  float s = b3s[col];
  for (int k = 0; k < 128; ++k)
    s += (float)x2[row * 128 + k] * w3s[k * 2 + col];
  const int r = b * PPC + pt * 16 + row;
  if (col == 0) out_logits[r] = s;
  else          out_weights[r] = s;
}

// =====================================================================
// Small per-row MLP2 (hidden relu, linear out) for tiny dims
// =====================================================================
__global__ __launch_bounds__(256)
void k_small_mlp2(const float* __restrict__ in, int K, int HID, int OUT,
                  const float* __restrict__ w1, const float* __restrict__ b1,
                  const float* __restrict__ w2, const float* __restrict__ b2,
                  float* __restrict__ out) {
  __shared__ float sIn[512];
  __shared__ float sHid[512];
  const int r = blockIdx.x, t = threadIdx.x;
  for (int k = t; k < K; k += 256) sIn[k] = in[(size_t)r * K + k];
  __syncthreads();
  for (int j = t; j < HID; j += 256) {
    float s = b1[j];
    for (int k = 0; k < K; ++k) s += sIn[k] * w1[(size_t)k * HID + j];
    sHid[j] = fmaxf(s, 0.0f);
  }
  __syncthreads();
  for (int o = t; o < OUT; o += 256) {
    float s = b2[o];
    for (int k = 0; k < HID; ++k) s += sHid[k] * w2[(size_t)k * OUT + o];
    out[(size_t)r * OUT + o] = s;
  }
}

// thread-per-output linear (edge head layer 2, N=4)
__global__ void k_lin_small(const float* __restrict__ in, int K, int N,
                            const float* __restrict__ w, const float* __restrict__ b,
                            float* __restrict__ out, int total) {
  const int i = blockIdx.x * blockDim.x + threadIdx.x;
  if (i >= total) return;
  const int r = i / N, n = i - r * N;
  float s = b[n];
  const float* row = in + (size_t)r * K;
  for (int k = 0; k < K; ++k) s += row[k] * w[k * N + n];
  out[i] = s;
}

__global__ void k_zero(float* __restrict__ p, int n) {
  const int i = blockIdx.x * blockDim.x + threadIdx.x;
  if (i < n) p[i] = 0.0f;
}

// segment-sum: agg[dst[e]] += msg[e]
__global__ void k_scatter_agg(const float* __restrict__ msg, const int* __restrict__ dst,
                              float* __restrict__ agg) {
  const int i = blockIdx.x * blockDim.x + threadIdx.x;   // EE*HDIM threads
  const int e = i >> 8, c = i & 255;
  atomicAdd(&agg[(size_t)dst[e] * HDIM + c], msg[i]);
}

// h = LN(h + upd) * g + b   (blockDim == HDIM == 256)
__global__ __launch_bounds__(256)
void k_ln(float* __restrict__ h, const float* __restrict__ upd,
          const float* __restrict__ g, const float* __restrict__ bb) {
  __shared__ float red[256];
  __shared__ float mu_s, rs_s;
  const int r = blockIdx.x, t = threadIdx.x;
  const float x = h[(size_t)r * 256 + t] + upd[(size_t)r * 256 + t];
  red[t] = x; __syncthreads();
  for (int s = 128; s > 0; s >>= 1) { if (t < s) red[t] += red[t + s]; __syncthreads(); }
  if (t == 0) mu_s = red[0] * (1.0f / 256.0f);
  __syncthreads();
  const float d = x - mu_s;
  red[t] = d * d; __syncthreads();
  for (int s = 128; s > 0; s >>= 1) { if (t < s) red[t] += red[t + s]; __syncthreads(); }
  if (t == 0) rs_s = rsqrtf(red[0] * (1.0f / 256.0f) + 1e-5f);
  __syncthreads();
  h[(size_t)r * 256 + t] = d * rs_s * g[t] + bb[t];
}

// ground-truth adjacency scatter / gather
__global__ void k_gt_scatter(const int* __restrict__ src, const int* __restrict__ dst,
                             const float* __restrict__ ea, float* __restrict__ gte,
                             float* __restrict__ gtw) {
  const int e = blockIdx.x * blockDim.x + threadIdx.x;
  if (e >= EE) return;
  const int s = src[e], d = dst[e];
  const int g = s >> 6, ls = s & 63, ld = d & 63;
  const int idx = g * (NPGC * NPGC) + ls * NPGC + ld;
  gte[idx] = 1.0f;
  gtw[idx] = ea[(size_t)e * 4];
}

__global__ void k_gt_gather(const float* __restrict__ gte, const float* __restrict__ gtw,
                            float* __restrict__ oe, float* __restrict__ ow) {
  const int i = blockIdx.x * blockDim.x + threadIdx.x;
  if (i >= BB * PPC) return;
  const int b = i / PPC, p = i - b * PPC;
  const int ii = p / 63, kk = p % 63;
  const int jj = kk + (kk >= ii ? 1 : 0);
  const int idx = b * (NPGC * NPGC) + ii * NPGC + jj;
  oe[i] = gte[idx];
  ow[i] = gtw[idx];
}

// =====================================================================
static inline int gemm_blocks(int M) { return ((M / 16) * 16 + 3) / 4; }

static inline void transpose(hipStream_t s, const float* W, float* WT, int K, int N) {
  k_transpose<<<(K * N + 255) / 256, 256, 0, s>>>(W, WT, K, N);
}

extern "C" void kernel_launch(void* const* d_in, const int* in_sizes, int n_in,
                              void* d_out, int out_size, void* d_ws, size_t ws_size,
                              hipStream_t stream) {
  (void)in_sizes; (void)n_in; (void)out_size; (void)ws_size;
  const float* z   = (const float*)d_in[0];
  const int*   ei  = (const int*)d_in[1];
  const float* ea  = (const float*)d_in[2];
  const int* src = ei;
  const int* dst = ei + EE;

  const float* node_q = (const float*)d_in[3];
  const float* zp_w1  = (const float*)d_in[4];
  const float* zp_b1  = (const float*)d_in[5];
  const float* zp_w2  = (const float*)d_in[6];
  const float* zp_b2  = (const float*)d_in[7];
  const float* cb_w1  = (const float*)d_in[8];
  const float* cb_b1  = (const float*)d_in[9];
  const float* cb_w2  = (const float*)d_in[10];
  const float* cb_b2  = (const float*)d_in[11];
  const float* mp_msg_w = (const float*)d_in[12];
  const float* mp_up_w1 = (const float*)d_in[13];
  const float* mp_up_b1 = (const float*)d_in[14];
  const float* mp_up_w2 = (const float*)d_in[15];
  const float* mp_up_b2 = (const float*)d_in[16];
  const float* ln_g   = (const float*)d_in[17];
  const float* ln_b   = (const float*)d_in[18];
  const float* cen_w1 = (const float*)d_in[19];
  const float* cen_b1 = (const float*)d_in[20];
  const float* cen_w2 = (const float*)d_in[21];
  const float* cen_b2 = (const float*)d_in[22];
  const float* mat_w1 = (const float*)d_in[23];
  const float* mat_b1 = (const float*)d_in[24];
  const float* mat_w2 = (const float*)d_in[25];
  const float* mat_b2 = (const float*)d_in[26];
  const float* nt_w1  = (const float*)d_in[27];
  const float* nt_b1  = (const float*)d_in[28];
  const float* nt_w2  = (const float*)d_in[29];
  const float* nt_b2  = (const float*)d_in[30];
  const float* fr_w1  = (const float*)d_in[31];
  const float* fr_b1  = (const float*)d_in[32];
  const float* fr_w2  = (const float*)d_in[33];
  const float* fr_b2  = (const float*)d_in[34];
  const float* ea_w1  = (const float*)d_in[35];
  const float* ea_b1  = (const float*)d_in[36];
  const float* ea_w2  = (const float*)d_in[37];
  const float* ea_b2  = (const float*)d_in[38];
  const float* adj_w1 = (const float*)d_in[39];
  const float* adj_b1 = (const float*)d_in[40];
  const float* adj_w2 = (const float*)d_in[41];
  const float* adj_b2 = (const float*)d_in[42];
  const float* adj_w3 = (const float*)d_in[43];
  const float* adj_b3 = (const float*)d_in[44];

  float* out = (float*)d_out;

  // ---- workspace layout (floats) ----
  float* ws  = (float*)d_ws;
  float* zc  = ws;                        // 64*256
  float* h   = zc  + BB * HDIM;           // NN*HDIM
  float* t1  = h   + NN * HDIM;           // NN*HDIM
  float* agg = t1  + NN * HDIM;           // NN*HDIM
  float* upd = agg + NN * HDIM;           // NN*HDIM
  float* msg = upd + NN * HDIM;           // EE*HDIM (reused as edge hidden)
  float* gte = msg + (size_t)EE * HDIM;   // B*64*64
  float* gtw = gte + NN * NPGC;           // B*64*64
  // transposed weights
  float* wt_cb1  = gtw + NN * NPGC;       // 256*512
  float* wt_cb2  = wt_cb1 + 512 * 256;    // 256*256
  float* wt_msg  = wt_cb2 + 256 * 256;    // 4 * 256*256
  float* wt_up1  = wt_msg + 4 * 256 * 256;
  float* wt_up2  = wt_up1 + 4 * 256 * 256;
  float* wt_ea1  = wt_up2 + 4 * 256 * 256; // 256*512
  float* wt_adj1 = wt_ea1 + 512 * 256;     // 256*768
  float* wt_adj2 = wt_adj1 + 768 * 256;    // 128*256

  // 0) transpose every WMMA-consumed weight matrix (cheap, < 6 MB total)
  transpose(stream, cb_w1, wt_cb1, 512, 256);
  transpose(stream, cb_w2, wt_cb2, 256, 256);
  for (int l = 0; l < LNUM; ++l) {
    transpose(stream, mp_msg_w + (size_t)l * 65536, wt_msg + (size_t)l * 65536, 256, 256);
    transpose(stream, mp_up_w1 + (size_t)l * 65536, wt_up1 + (size_t)l * 65536, 256, 256);
    transpose(stream, mp_up_w2 + (size_t)l * 65536, wt_up2 + (size_t)l * 65536, 256, 256);
  }
  transpose(stream, ea_w1, wt_ea1, 512, 256);
  transpose(stream, adj_w1, wt_adj1, 768, 256);
  transpose(stream, adj_w2, wt_adj2, 256, 128);

  // 1) z_cond = mlp2(z)   (64 rows, 128->512->256)
  k_small_mlp2<<<BB, 256, 0, stream>>>(z, ZD, 2 * HDIM, HDIM,
                                       zp_w1, zp_b1, zp_w2, zp_b2, zc);

  // 2) h = mlp2(concat(q, z_cond))   (4096 x 512 -> 256 relu -> 256)
  k_gemm<2><<<gemm_blocks(NN), 128, 0, stream>>>(
      node_q, zc, nullptr, nullptr, 256, wt_cb1, cb_b1, t1, NN, 512, 1);
  k_gemm<0><<<gemm_blocks(NN), 128, 0, stream>>>(
      t1, nullptr, nullptr, nullptr, 256, wt_cb2, cb_b2, h, NN, 256, 0);

  // 3) message passing, L = 4
  for (int l = 0; l < LNUM; ++l) {
    k_gemm<1><<<gemm_blocks(EE), 128, 0, stream>>>(
        h, h, src, src, 256, wt_msg + (size_t)l * 65536, nullptr, msg, EE, 256, 0);
    k_zero<<<(NN * HDIM + 255) / 256, 256, 0, stream>>>(agg, NN * HDIM);
    k_scatter_agg<<<(EE * HDIM) / 256, 256, 0, stream>>>(msg, dst, agg);
    k_gemm<0><<<gemm_blocks(NN), 128, 0, stream>>>(
        agg, nullptr, nullptr, nullptr, 256, wt_up1 + (size_t)l * 65536,
        mp_up_b1 + l * HDIM, t1, NN, 256, 1);
    k_gemm<0><<<gemm_blocks(NN), 128, 0, stream>>>(
        t1, nullptr, nullptr, nullptr, 256, wt_up2 + (size_t)l * 65536,
        mp_up_b2 + l * HDIM, upd, NN, 256, 0);
    k_ln<<<NN, 256, 0, stream>>>(h, upd, ln_g + l * HDIM, ln_b + l * HDIM);
  }

  // 4) edge-attr head: concat(h[src],h[dst]) @ ea_w1 relu -> @ ea_w2 (N=4)
  k_gemm<1><<<gemm_blocks(EE), 128, 0, stream>>>(
      h, h, src, dst, 256, wt_ea1, ea_b1, msg, EE, 512, 1);
  k_lin_small<<<(EE * 4 + 255) / 256, 256, 0, stream>>>(
      msg, HDIM, 4, ea_w2, ea_b2, out + OFF_EA, EE * 4);

  // 5) ground-truth adjacency targets
  k_zero<<<(2 * NN * NPGC + 255) / 256, 256, 0, stream>>>(gte, 2 * NN * NPGC);
  k_gt_scatter<<<(EE + 255) / 256, 256, 0, stream>>>(src, dst, ea, gte, gtw);
  k_gt_gather<<<(BB * PPC + 255) / 256, 256, 0, stream>>>(gte, gtw,
                                                          out + OFF_TE, out + OFF_TW);

  // 6) per-node heads
  k_small_mlp2<<<NN, 256, 0, stream>>>(h, HDIM, HDIM / 2, 3,
                                       cen_w1, cen_b1, cen_w2, cen_b2, out + OFF_CEN);
  k_small_mlp2<<<NN, 256, 0, stream>>>(h, HDIM, HDIM, 32,
                                       mat_w1, mat_b1, mat_w2, mat_b2, out + OFF_MAT);
  k_small_mlp2<<<NN, 256, 0, stream>>>(h, HDIM, HDIM / 2, 3,
                                       nt_w1, nt_b1, nt_w2, nt_b2, out + OFF_NT);
  k_small_mlp2<<<NN, 256, 0, stream>>>(h, HDIM, HDIM / 4, 1,
                                       fr_w1, fr_b1, fr_w2, fr_b2, out + OFF_FR);

  // 7) fused pair-adjacency MLP (dominant GEMM chain), one wave / 16 pairs
  k_adj_fused<<<BB * (PPC / 16), 32, 0, stream>>>(
      h, wt_adj1, adj_b1, wt_adj2, adj_b2, adj_w3, adj_b3,
      out + OFF_AL, out + OFF_AW);
}
